// MultiHeadSelfAttention_68779606278652
// MI455X (gfx1250) — matrix operand order repaired and checked by reading, not compiled, not run on hardware
//
#include <hip/hip_runtime.h>

// MI455X / gfx1250: wave32, WMMA bf16 16x16x32 with f32 accumulate.
// MHA forward: B=1, T=4096, C=1024, H=16, Dh=64.
// Workspace layout (assumes ws_size >= 48 MB):
//   Xb(8MB) Wqb(2) Wkb(2) Wvb(2) Wob(2) Qb(8) Kb(8) Vtb(8) Ab(8)

typedef __bf16 bf16_t;
typedef __attribute__((ext_vector_type(16))) __bf16 v16bf;
typedef __attribute__((ext_vector_type(8)))  float v8f;
typedef __attribute__((ext_vector_type(4)))  unsigned int v4u;
typedef __attribute__((ext_vector_type(4)))  int v4i;

#define SEQ_T  4096
#define DMODEL 1024
#define DHEAD  64

#if __has_builtin(__builtin_amdgcn_global_load_async_to_lds_b128) && \
    __has_builtin(__builtin_amdgcn_s_wait_asynccnt)
#define USE_ASYNC 1
#else
#define USE_ASYNC 0
#endif

// ---- WMMA fragment helpers (layouts per CDNA5 ISA 7.12.2, wave32) ----------

// A-matrix 16x32 bf16: lane m = lane&15, half hi = lane>>4.
// element e: k = 16*(e/8) + 8*hi + (e&7)  -> two contiguous 16B loads.
static __device__ __forceinline__ v16bf load_frag_a(const bf16_t* __restrict__ p0,
                                                    int ld, int lane) {
  const int m  = lane & 15;
  const int hi = lane >> 4;
  const bf16_t* p = p0 + (size_t)m * ld + hi * 8;
  union { v4u u[2]; v16bf v; } f;
  f.u[0] = *(const v4u*)(p);
  f.u[1] = *(const v4u*)(p + 16);
  return f.v;
}

// B-matrix 32x16 bf16 loaded from B^T (row-major [n][k]): lane n = lane&15,
// element e: k = 16*(e/8) + 8*hi + (e&7) -> two contiguous 16B loads.
static __device__ __forceinline__ v16bf load_frag_bt(const bf16_t* __restrict__ p0,
                                                     int ld, int lane) {
  const int n  = lane & 15;
  const int hi = lane >> 4;
  const bf16_t* p = p0 + (size_t)n * ld + hi * 8;
  union { v4u u[2]; v16bf v; } f;
  f.u[0] = *(const v4u*)(p);
  f.u[1] = *(const v4u*)(p + 16);
  return f.v;
}

static __device__ __forceinline__ v8f wmma_bf16(v16bf a, v16bf b, v8f c) {
  // (neg_a, A, neg_b, B, c_mod, C, reuse_a, reuse_b)
  return __builtin_amdgcn_wmma_f32_16x16x32_bf16(false, a, false, b, (short)0, c,
                                                 false, false);
}

// ---- f32 -> bf16 conversion ------------------------------------------------

__global__ void cvt_f32_to_bf16(const float* __restrict__ src,
                                bf16_t* __restrict__ dst, int n) {
  int i = blockIdx.x * blockDim.x + threadIdx.x;
  const int stride = gridDim.x * blockDim.x;
  for (; i < n; i += stride) dst[i] = (bf16_t)src[i];
}

// ---- GEMM: Y[M,N] = A[M,K] * Bt[N,K]^T (bf16 in, f32 accum) ----------------
// Block = 8 waves sharing one 64-col B strip (double-buffered in LDS via
// async-to-LDS when available); each wave owns a 32x64 output tile.
// MODE 0: bf16 row-major [M,N]; MODE 1: bf16 transposed [N,M] (for V);
// MODE 2: f32 row-major [M,N].

// Stage a 64(n) x 32(k) tile of B^T into LDS row-major [64][32] (4 KB).
// 256 threads: thread t covers n = t/4, elements seg..seg+7 (one B128 each).
static __device__ __forceinline__ void stage_b(const bf16_t* __restrict__ Bt,
                                               int cb, int kk, int K,
                                               bf16_t* dst, int tid) {
  const int n   = tid >> 2;
  const int seg = (tid & 3) * 8;
  const bf16_t* src = Bt + (size_t)(cb + n) * K + kk + seg;
  bf16_t* d = dst + n * 32 + seg;
#if USE_ASYNC
  __builtin_amdgcn_global_load_async_to_lds_b128((v4i*)src, (v4i*)d, 0, 0);
#else
  *(v4u*)d = *(const v4u*)src;
#endif
}

template <int MODE>
static __device__ __forceinline__ void store_tile(void* __restrict__ Out,
                                                  const v8f* acc, int m0, int cb,
                                                  int M, int N, int lane) {
  const int n = lane & 15, hi = lane >> 4;
  if (MODE == 0) {
    bf16_t* out = (bf16_t*)Out;
#pragma unroll
    for (int j = 0; j < 4; ++j)
#pragma unroll
      for (int i = 0; i < 8; ++i)
        out[(size_t)(m0 + i + 8 * hi) * N + cb + j * 16 + n] = (bf16_t)acc[j][i];
  } else if (MODE == 1) {
    bf16_t* outT = (bf16_t*)Out;
#pragma unroll
    for (int j = 0; j < 4; ++j) {
      union { v4u u; bf16_t h[8]; } st;
#pragma unroll
      for (int i = 0; i < 8; ++i) st.h[i] = (bf16_t)acc[j][i];
      *(v4u*)(outT + (size_t)(cb + j * 16 + n) * M + m0 + 8 * hi) = st.u;
    }
  } else {
    float* out = (float*)Out;
#pragma unroll
    for (int j = 0; j < 4; ++j)
#pragma unroll
      for (int i = 0; i < 8; ++i)
        out[(size_t)(m0 + i + 8 * hi) * N + cb + j * 16 + n] = acc[j][i];
  }
}

template <int MODE>
__global__ __launch_bounds__(256) void gemm_bf16(const bf16_t* __restrict__ A,
                                                 const bf16_t* __restrict__ Bt,
                                                 void* __restrict__ Out,
                                                 int M, int N, int K) {
  __shared__ __align__(16) bf16_t bstage[2][64 * 32];  // 2 x 4 KB
  const int tid  = threadIdx.x;
  const int lane = tid & 31;
  const int wid  = tid >> 5;
  const int nblk = N >> 6;
  const int bn   = blockIdx.x % nblk;
  const int bm   = blockIdx.x / nblk;
  const int cb   = bn * 64;
  const int m0   = bm * 256 + wid * 32;

  v8f acc0[4] = {}, acc1[4] = {};
  const bf16_t* ap = A + (size_t)m0 * K;

  stage_b(Bt, cb, 0, K, &bstage[0][0], tid);
  int buf = 0;
  for (int kk = 0; kk < K; kk += 32, buf ^= 1) {
    if (kk + 32 < K) {
      stage_b(Bt, cb, kk + 32, K, &bstage[buf ^ 1][0], tid);
#if USE_ASYNC
      __builtin_amdgcn_s_wait_asynccnt(1);  // current buffer's loads done
#endif
    } else {
#if USE_ASYNC
      __builtin_amdgcn_s_wait_asynccnt(0);
#endif
    }
    __syncthreads();
    __builtin_prefetch(ap + kk + 64, 0, 1);  // global_prefetch for A stream
    const v16bf a0 = load_frag_a(ap + kk, K, lane);
    const v16bf a1 = load_frag_a(ap + (size_t)16 * K + kk, K, lane);
#pragma unroll
    for (int j = 0; j < 4; ++j) {
      const v16bf b = load_frag_bt(&bstage[buf][j * 16 * 32], 32, lane);
      acc0[j] = wmma_bf16(a0, b, acc0[j]);
      acc1[j] = wmma_bf16(a1, b, acc1[j]);
    }
    __syncthreads();
  }
  store_tile<MODE>(Out, acc0, m0, cb, M, N, lane);
  store_tile<MODE>(Out, acc1, m0 + 16, cb, M, N, lane);
}

// ---- Causal flash attention (one wave per head x 16-query tile) ------------
// Q,K row-major [T, DMODEL] (head slice at h*64), Vt transposed [DMODEL, T].
// 64-key blocks: 8 WMMA for QK^T, one online-softmax pass, 8 WMMA for PV.

__global__ __launch_bounds__(256) void attn_fwd(const bf16_t* __restrict__ Q,
                                                const bf16_t* __restrict__ Kmat,
                                                const bf16_t* __restrict__ Vt,
                                                bf16_t* __restrict__ Aout) {
  __shared__ __align__(16) bf16_t ptile[8][16 * 64];  // 2 KB per wave
  const int lane = threadIdx.x & 31;
  const int wid  = threadIdx.x >> 5;
  const int h    = blockIdx.y;
  const int q0   = (blockIdx.x * 8 + wid) * 16;
  const int ml   = lane & 15;
  const int hi   = lane >> 4;

  const bf16_t* qp = Q + (size_t)q0 * DMODEL + h * DHEAD;
  const v16bf aq0 = load_frag_a(qp, DMODEL, lane);
  const v16bf aq1 = load_frag_a(qp + 32, DMODEL, lane);

  float mrow[8], lrow[8];
#pragma unroll
  for (int i = 0; i < 8; ++i) { mrow[i] = -1e30f; lrow[i] = 0.0f; }
  v8f o[4] = {};

  const float sc = 0.125f;   // 1/sqrt(64)
  const int kend = q0 + 16;  // causal bound

  for (int kb = 0; kb < kend; kb += 64) {
    // S = Q (16x64) * K^T (64x64) -> four 16x16 C tiles
    v8f s[4] = {};
#pragma unroll
    for (int nt = 0; nt < 4; ++nt) {
      const bf16_t* kp = Kmat + (size_t)(kb + nt * 16) * DMODEL + h * DHEAD;
      s[nt] = wmma_bf16(aq0, load_frag_bt(kp, DMODEL, lane), s[nt]);
      s[nt] = wmma_bf16(aq1, load_frag_bt(kp + 32, DMODEL, lane), s[nt]);
    }

    // Online softmax. C layout: row = i + 8*hi, col = nt*16 + ml.
    float oscale[8];
#pragma unroll
    for (int i = 0; i < 8; ++i) {
      const int t = q0 + i + 8 * hi;
      float v[4];
#pragma unroll
      for (int nt = 0; nt < 4; ++nt) {
        const int key = kb + nt * 16 + ml;
        v[nt] = (key <= t) ? s[nt][i] * sc : -1e30f;
      }
      float mx = fmaxf(fmaxf(v[0], v[1]), fmaxf(v[2], v[3]));
#pragma unroll
      for (int d = 1; d <= 8; d <<= 1) mx = fmaxf(mx, __shfl_xor(mx, d, 32));
      const float mnew = fmaxf(mrow[i], mx);
      const float scal = __expf(mrow[i] - mnew);
      float ps = 0.0f;
#pragma unroll
      for (int nt = 0; nt < 4; ++nt) {
        const float p = __expf(v[nt] - mnew);
        ps += p;
        ptile[wid][(i + 8 * hi) * 64 + nt * 16 + ml] = (bf16_t)p;
      }
#pragma unroll
      for (int d = 1; d <= 8; d <<= 1) ps += __shfl_xor(ps, d, 32);
      lrow[i] = lrow[i] * scal + ps;
      mrow[i] = mnew;
      oscale[i] = scal;
    }
#pragma unroll
    for (int j = 0; j < 4; ++j)
#pragma unroll
      for (int i = 0; i < 8; ++i) o[j][i] *= oscale[i];

    // same-wave LDS store->load re-layout (C layout -> A layout)
    asm volatile("s_wait_dscnt 0x0" ::: "memory");
    const v16bf ap0 = load_frag_a(&ptile[wid][0], 64, lane);        // keys 0..31
    const v16bf ap1 = load_frag_a(&ptile[wid][0] + 32, 64, lane);   // keys 32..63
#pragma unroll
    for (int j = 0; j < 4; ++j) {
      const bf16_t* vp = Vt + (size_t)(h * DHEAD + j * 16) * SEQ_T + kb;
      o[j] = wmma_bf16(ap0, load_frag_bt(vp, SEQ_T, lane), o[j]);
      o[j] = wmma_bf16(ap1, load_frag_bt(vp + 32, SEQ_T, lane), o[j]);
    }
  }

  bf16_t* outp = Aout + (size_t)q0 * DMODEL + h * DHEAD;
#pragma unroll
  for (int j = 0; j < 4; ++j)
#pragma unroll
    for (int i = 0; i < 8; ++i) {
      const float val = o[j][i] / lrow[i];
      outp[(size_t)(i + 8 * hi) * DMODEL + j * 16 + ml] = (bf16_t)val;
    }
}

// ---- host launcher ---------------------------------------------------------

extern "C" void kernel_launch(void* const* d_in, const int* in_sizes, int n_in,
                              void* d_out, int out_size, void* d_ws, size_t ws_size,
                              hipStream_t stream) {
  (void)in_sizes; (void)n_in; (void)out_size; (void)ws_size;
  const float* x  = (const float*)d_in[0];
  const float* Wq = (const float*)d_in[1];
  const float* Wk = (const float*)d_in[2];
  const float* Wv = (const float*)d_in[3];
  const float* Wo = (const float*)d_in[4];

  char* ws = (char*)d_ws;
  const size_t szX = (size_t)SEQ_T * DMODEL * sizeof(bf16_t);   // 8 MB
  const size_t szW = (size_t)DMODEL * DMODEL * sizeof(bf16_t);  // 2 MB
  bf16_t* Xb  = (bf16_t*)ws; ws += szX;
  bf16_t* Wqb = (bf16_t*)ws; ws += szW;
  bf16_t* Wkb = (bf16_t*)ws; ws += szW;
  bf16_t* Wvb = (bf16_t*)ws; ws += szW;
  bf16_t* Wob = (bf16_t*)ws; ws += szW;
  bf16_t* Qb  = (bf16_t*)ws; ws += szX;
  bf16_t* Kb  = (bf16_t*)ws; ws += szX;
  bf16_t* Vtb = (bf16_t*)ws; ws += szX;
  bf16_t* Ab  = (bf16_t*)ws; ws += szX;

  const int nX = SEQ_T * DMODEL;
  const int nW = DMODEL * DMODEL;
  cvt_f32_to_bf16<<<2048, 256, 0, stream>>>(x,  Xb,  nX);
  cvt_f32_to_bf16<<<1024, 256, 0, stream>>>(Wq, Wqb, nW);
  cvt_f32_to_bf16<<<1024, 256, 0, stream>>>(Wk, Wkb, nW);
  cvt_f32_to_bf16<<<1024, 256, 0, stream>>>(Wv, Wvb, nW);
  cvt_f32_to_bf16<<<1024, 256, 0, stream>>>(Wo, Wob, nW);

  // grid: (N/64) * (M/256) = 16 * 16 = 256 blocks of 8 waves
  gemm_bf16<0><<<256, 256, 0, stream>>>(Xb, Wqb, (void*)Qb,  SEQ_T, DMODEL, DMODEL);
  gemm_bf16<0><<<256, 256, 0, stream>>>(Xb, Wkb, (void*)Kb,  SEQ_T, DMODEL, DMODEL);
  gemm_bf16<1><<<256, 256, 0, stream>>>(Xb, Wvb, (void*)Vtb, SEQ_T, DMODEL, DMODEL);

  attn_fwd<<<dim3(32, 16), 256, 0, stream>>>(Qb, Kb, Vtb, Ab);

  gemm_bf16<2><<<256, 256, 0, stream>>>(Ab, Wob, d_out, SEQ_T, DMODEL, DMODEL);
}
